// SetUpdate_rec2_flow_78426102825599
// MI455X (gfx1250) — matrix operand order repaired and checked by reading, not compiled, not run on hardware
//
#include <hip/hip_runtime.h>
#include <hip/hip_bf16.h>

typedef __attribute__((ext_vector_type(16))) _Float16 v16h;
typedef __attribute__((ext_vector_type(8)))  float    v8f;

#define BATCH 2
#define NPTS  4096
#define KNB   16

// ---------------------------------------------------------------------------
// helpers
// ---------------------------------------------------------------------------
__device__ __forceinline__ v8f zero8() {
  v8f z = {0.f, 0.f, 0.f, 0.f, 0.f, 0.f, 0.f, 0.f};
  return z;
}

// B-panel swizzle: channel-in-panel c (K dim of WMMA), column col (N dim).
// Panel layout per 32-channel chunk: 32 lanes x 16 halves, lane reads
// contiguous 32B. lanes 0-15 hold kin 0..15 (col = lane), lanes 16-31 hold
// kin 16..31.
__device__ __forceinline__ int bswz(int c, int col) {
  int chunk = c >> 5;
  int kin   = c & 31;
  return chunk * 512 + (col + ((kin >> 4) << 4)) * 16 + (kin & 15);
}

// D = W * X ; W panel pre-swizzled in global (A fragments contiguous per
// (tile,chunk)), X panel swizzled in LDS. One wave does the whole tile set.
template <int CHUNKS, int TILES>
__device__ __forceinline__ void wmma_gemm(const _Float16* __restrict__ wp,
                                          const _Float16* xp, int lane,
                                          v8f* acc) {
#pragma unroll
  for (int t = 0; t < TILES; ++t) acc[t] = zero8();
#pragma unroll
  for (int ch = 0; ch < CHUNKS; ++ch) {
    v16h bf = *(const v16h*)(xp + ch * 512 + lane * 16);
#pragma unroll
    for (int t = 0; t < TILES; ++t) {
      v16h af = *(const v16h*)(wp + ((size_t)(t * CHUNKS + ch) * 32 + lane) * 16);
      acc[t] = __builtin_amdgcn_wmma_f32_16x16x32_f16(
          false, af, false, bf, (short)0, acc[t], false, false);
    }
  }
}

// max over the 16 columns of each D row (columns live across a 16-lane half)
template <int T>
__device__ __forceinline__ void rowmax16(v8f* acc) {
#pragma unroll
  for (int t = 0; t < T; ++t)
#pragma unroll
    for (int v = 0; v < 8; ++v) {
      float x = acc[t][v];
      x = fmaxf(x, __shfl_xor(x, 1, 32));
      x = fmaxf(x, __shfl_xor(x, 2, 32));
      x = fmaxf(x, __shfl_xor(x, 4, 32));
      x = fmaxf(x, __shfl_xor(x, 8, 32));
      acc[t][v] = x;
    }
}

__device__ __forceinline__ float leaky(float x) {
  return x >= 0.f ? x : 0.1f * x;
}

// ---------------------------------------------------------------------------
// weight prep: f32 [Cout,Cin] -> swizzled f16 A-panel (zero padded channels)
// panel index e = ((tile*chunks + chunk)*32 + lane)*16 + slot
// ---------------------------------------------------------------------------
__global__ void prep_w_kernel(const float* __restrict__ W,
                              _Float16* __restrict__ panel, int Cin, int chunks,
                              int total) {
  int e = blockIdx.x * 256 + threadIdx.x;
  if (e >= total) return;
  int slot  = e & 15;
  int lane  = (e >> 4) & 31;
  int rest  = e >> 9;
  int chunk = rest % chunks;
  int tile  = rest / chunks;
  int m = lane & 15, g = lane >> 4;
  int base = slot;   // contiguous k: kin = 16*g + slot (matches B panel)
  int kin  = base + 16 * g;
  int c = chunk * 32 + kin;
  int o = tile * 16 + m;
  float v = (c < Cin) ? W[(size_t)o * Cin + c] : 0.f;
  panel[e] = (_Float16)v;
}

// ---------------------------------------------------------------------------
// brute-force KNN: one thread per query, register top-16 insertion sort
// ---------------------------------------------------------------------------
__global__ void knn_kernel(const float* __restrict__ q,
                           const float* __restrict__ t, int* __restrict__ idx) {
  int p = blockIdx.x * 256 + threadIdx.x;
  int b = p >> 12, n = p & 4095;
  const float* qb = q + (size_t)b * 3 * NPTS;
  const float* tb = t + (size_t)b * 3 * NPTS;
  float qx = qb[n], qy = qb[NPTS + n], qz = qb[2 * NPTS + n];
  float bd[KNB];
  int   bi[KNB];
#pragma unroll
  for (int i = 0; i < KNB; ++i) { bd[i] = 3.4e38f; bi[i] = 0; }
  for (int m = 0; m < NPTS; ++m) {
    float dx = qx - tb[m];
    float dy = qy - tb[NPTS + m];
    float dz = qz - tb[2 * NPTS + m];
    float d = dx * dx + dy * dy + dz * dz;
    if (d < bd[KNB - 1]) {
      bd[KNB - 1] = d; bi[KNB - 1] = m;
#pragma unroll
      for (int i = KNB - 1; i > 0; --i) {
        if (bd[i] < bd[i - 1]) {
          float td = bd[i]; bd[i] = bd[i - 1]; bd[i - 1] = td;
          int   ti = bi[i]; bi[i] = bi[i - 1]; bi[i - 1] = ti;
        }
      }
    }
  }
#pragma unroll
  for (int i = 0; i < KNB; ++i) idx[(size_t)p * KNB + i] = bi[i];
}

// ---------------------------------------------------------------------------
// flow conv1: gather [flow0, dxyz] (6ch -> pad 32) -> 6->64 conv, raw+bias to h
// h layout: [b*N+n][c=64][k=16] f32
// ---------------------------------------------------------------------------
__global__ void flow_conv1_kernel(const float* __restrict__ flow0,
                                  const float* __restrict__ xyz0,
                                  const int* __restrict__ idx0,
                                  const _Float16* __restrict__ pw,
                                  const float* __restrict__ bias,
                                  float* __restrict__ h) {
  __shared__ alignas(32) _Float16 X[8][512];
  __shared__ int   sidx[8][16];
  __shared__ float scen[8][3];
  int lane = threadIdx.x & 31, wave = threadIdx.x >> 5;
  int p0 = blockIdx.x * 8;
  if (threadIdx.x < 128) {
    int s = threadIdx.x >> 4, k = threadIdx.x & 15;
    sidx[s][k] = idx0[(size_t)(p0 + s) * KNB + k];
  } else if (threadIdx.x < 152) {
    int r = threadIdx.x - 128; int s = r / 3, c = r % 3;
    int pp = p0 + s;
    scen[s][c] = xyz0[((size_t)(pp >> 12) * 3 + c) * NPTS + (pp & 4095)];
  }
  __syncthreads();
  for (int e = threadIdx.x; e < 8 * 32 * 16; e += 256) {
    int col = e & 15, c = (e >> 4) & 31, s = e >> 9;
    int pp = p0 + s; int b = pp >> 12;
    int j = sidx[s][col];
    float v = 0.f;
    if (c < 3)      v = flow0[((size_t)b * 3 + c) * NPTS + j];
    else if (c < 6) v = xyz0[((size_t)b * 3 + (c - 3)) * NPTS + j] - scen[s][c - 3];
    X[s][bswz(c, col)] = (_Float16)v;
  }
  __syncthreads();
  int pp = p0 + wave;
  v8f acc[4];
  wmma_gemm<1, 4>(pw, X[wave], lane, acc);
  int col = lane & 15, g = lane >> 4;
  float* hb = h + (size_t)pp * 1024;
#pragma unroll
  for (int t = 0; t < 4; ++t)
#pragma unroll
    for (int v = 0; v < 8; ++v) {
      int c = t * 16 + v + 8 * g;
      hb[c * 16 + col] = acc[t][v] + bias[c];
    }
}

// ---------------------------------------------------------------------------
// group-norm stats: one block per (b, group of 16 ch); mean + inv_std
// ---------------------------------------------------------------------------
__global__ void gn_stats_kernel(const float* __restrict__ h,
                                float* __restrict__ stats) {
  int bg = blockIdx.x; int b = bg >> 2; int g = bg & 3;
  float s = 0.f, s2 = 0.f;
  for (int i = threadIdx.x; i < NPTS * 16 * 16; i += 256) {
    int k = i & 15, c = (i >> 4) & 15, n = i >> 8;
    float v = h[(((size_t)b * NPTS + n) * 64 + g * 16 + c) * 16 + k];
    s += v; s2 += v * v;
  }
  __shared__ float rs[256], rq[256];
  rs[threadIdx.x] = s; rq[threadIdx.x] = s2;
  __syncthreads();
  for (int st = 128; st > 0; st >>= 1) {
    if (threadIdx.x < st) {
      rs[threadIdx.x] += rs[threadIdx.x + st];
      rq[threadIdx.x] += rq[threadIdx.x + st];
    }
    __syncthreads();
  }
  if (threadIdx.x == 0) {
    float cnt = (float)NPTS * 256.f;
    float m = rs[0] / cnt;
    float var = rq[0] / cnt - m * m;
    stats[bg * 2] = m;
    stats[bg * 2 + 1] = rsqrtf(var + 1e-5f);
  }
}

// ---------------------------------------------------------------------------
// flow conv2: normalize(h)+leaky -> 64->64 conv, write back in place (+bias)
// ---------------------------------------------------------------------------
__global__ void flow_conv2_kernel(float* __restrict__ h,
                                  const float* __restrict__ stats,
                                  const float* __restrict__ gam,
                                  const float* __restrict__ bet,
                                  const _Float16* __restrict__ pw,
                                  const float* __restrict__ bias) {
  __shared__ alignas(32) _Float16 X[8][1024];
  int lane = threadIdx.x & 31, wave = threadIdx.x >> 5;
  int p0 = blockIdx.x * 8;
  for (int e = threadIdx.x; e < 8 * 64 * 16; e += 256) {
    int col = e & 15, c = (e >> 4) & 63, s = e >> 10;
    int pp = p0 + s; int b = pp >> 12;
    float v = h[(size_t)pp * 1024 + c * 16 + col];
    int grp = b * 4 + (c >> 4);
    float x = (v - stats[grp * 2]) * stats[grp * 2 + 1] * gam[c] + bet[c];
    X[s][bswz(c, col)] = (_Float16)leaky(x);
  }
  __syncthreads();
  int pp = p0 + wave;
  v8f acc[4];
  wmma_gemm<2, 4>(pw, X[wave], lane, acc);
  int col = lane & 15, g = lane >> 4;
  float* hb = h + (size_t)pp * 1024;
#pragma unroll
  for (int t = 0; t < 4; ++t)
#pragma unroll
    for (int v = 0; v < 8; ++v) {
      int c = t * 16 + v + 8 * g;
      hb[c * 16 + col] = acc[t][v] + bias[c];
    }
}

// ---------------------------------------------------------------------------
// flow finalize: normalize+leaky+max over K -> flow_feat [B,64,N]
// ---------------------------------------------------------------------------
__global__ void flow_final_kernel(const float* __restrict__ h,
                                  const float* __restrict__ stats,
                                  const float* __restrict__ gam,
                                  const float* __restrict__ bet,
                                  float* __restrict__ ff) {
  int e = blockIdx.x * 256 + threadIdx.x;   // B*64*N
  int n = e & 4095, c = (e >> 12) & 63, b = e >> 18;
  int grp = b * 4 + (c >> 4);
  float m = stats[grp * 2], inv = stats[grp * 2 + 1];
  const float* hp = h + (((size_t)b * NPTS + n) * 64 + c) * 16;
  float mx = -3.4e38f;
#pragma unroll
  for (int k = 0; k < 16; ++k) {
    float x = (hp[k] - m) * inv * gam[c] + bet[c];
    mx = fmaxf(mx, leaky(x));
  }
  ff[((size_t)b * 64 + c) * NPTS + n] = mx;
}

// ---------------------------------------------------------------------------
// GRU z/r gates: gather [corr0, flow_feat, state, dxyz] (195 -> pad 224)
// conv 195->64, max over K, sigmoid
// ---------------------------------------------------------------------------
__global__ void gates_zr_kernel(const float* __restrict__ corr0,
                                const float* __restrict__ ff,
                                const float* __restrict__ state,
                                const float* __restrict__ xyz0,
                                const int* __restrict__ idx0,
                                const _Float16* __restrict__ pwz,
                                const float* __restrict__ bz,
                                const _Float16* __restrict__ pwr,
                                const float* __restrict__ br,
                                float* __restrict__ zg, float* __restrict__ rg) {
  __shared__ alignas(32) _Float16 X[8][3584];
  __shared__ int   sidx[8][16];
  __shared__ float scen[8][3];
  int lane = threadIdx.x & 31, wave = threadIdx.x >> 5;
  int p0 = blockIdx.x * 8;
  if (threadIdx.x < 128) {
    int s = threadIdx.x >> 4, k = threadIdx.x & 15;
    sidx[s][k] = idx0[(size_t)(p0 + s) * KNB + k];
  } else if (threadIdx.x < 152) {
    int r = threadIdx.x - 128; int s = r / 3, c = r % 3;
    int pp = p0 + s;
    scen[s][c] = xyz0[((size_t)(pp >> 12) * 3 + c) * NPTS + (pp & 4095)];
  }
  __syncthreads();
  for (int e = threadIdx.x; e < 8 * 224 * 16; e += 256) {
    int col = e & 15;
    int c = (e >> 4) % 224;
    int s = e / (224 * 16);
    int pp = p0 + s; int b = pp >> 12;
    int j = sidx[s][col];
    float v;
    if (c < 64)       v = corr0[((size_t)b * 64 + c) * NPTS + j];
    else if (c < 128) v = ff[((size_t)b * 64 + (c - 64)) * NPTS + j];
    else if (c < 192) v = state[((size_t)b * 64 + (c - 128)) * NPTS + j];
    else if (c < 195) v = xyz0[((size_t)b * 3 + (c - 192)) * NPTS + j] - scen[s][c - 192];
    else              v = 0.f;
    X[s][bswz(c, col)] = (_Float16)v;
  }
  __syncthreads();
  int pp = p0 + wave; int b = pp >> 12, n = pp & 4095;
  v8f acc[4];
  wmma_gemm<7, 4>(pwz, X[wave], lane, acc);
  rowmax16<4>(acc);
  if ((lane & 15) == 0) {
    int g = lane >> 4;
#pragma unroll
    for (int t = 0; t < 4; ++t)
#pragma unroll
      for (int v = 0; v < 8; ++v) {
        int c = t * 16 + v + 8 * g;
        float x = acc[t][v] + bz[c];
        zg[((size_t)b * 64 + c) * NPTS + n] = 1.f / (1.f + __expf(-x));
      }
  }
  wmma_gemm<7, 4>(pwr, X[wave], lane, acc);
  rowmax16<4>(acc);
  if ((lane & 15) == 0) {
    int g = lane >> 4;
#pragma unroll
    for (int t = 0; t < 4; ++t)
#pragma unroll
      for (int v = 0; v < 8; ++v) {
        int c = t * 16 + v + 8 * g;
        float x = acc[t][v] + br[c];
        rg[((size_t)b * 64 + c) * NPTS + n] = 1.f / (1.f + __expf(-x));
      }
  }
}

// ---------------------------------------------------------------------------
// GRU q gate + state update
// ---------------------------------------------------------------------------
__global__ void gate_q_kernel(const float* __restrict__ corr0,
                              const float* __restrict__ ff,
                              const float* __restrict__ state,
                              const float* __restrict__ xyz0,
                              const int* __restrict__ idx0,
                              const _Float16* __restrict__ pwq,
                              const float* __restrict__ bq,
                              const float* __restrict__ zg,
                              const float* __restrict__ rg,
                              float* __restrict__ nst) {
  __shared__ alignas(32) _Float16 X[8][3584];
  __shared__ int   sidx[8][16];
  __shared__ float scen[8][3];
  int lane = threadIdx.x & 31, wave = threadIdx.x >> 5;
  int p0 = blockIdx.x * 8;
  if (threadIdx.x < 128) {
    int s = threadIdx.x >> 4, k = threadIdx.x & 15;
    sidx[s][k] = idx0[(size_t)(p0 + s) * KNB + k];
  } else if (threadIdx.x < 152) {
    int r = threadIdx.x - 128; int s = r / 3, c = r % 3;
    int pp = p0 + s;
    scen[s][c] = xyz0[((size_t)(pp >> 12) * 3 + c) * NPTS + (pp & 4095)];
  }
  __syncthreads();
  for (int e = threadIdx.x; e < 8 * 224 * 16; e += 256) {
    int col = e & 15;
    int c = (e >> 4) % 224;
    int s = e / (224 * 16);
    int pp = p0 + s; int b = pp >> 12;
    int j = sidx[s][col];
    float v;
    if (c < 64)       v = corr0[((size_t)b * 64 + c) * NPTS + j];
    else if (c < 128) v = ff[((size_t)b * 64 + (c - 64)) * NPTS + j];
    else if (c < 192) {
      size_t o = ((size_t)b * 64 + (c - 128)) * NPTS + j;
      v = rg[o] * state[o];
    } else if (c < 195) v = xyz0[((size_t)b * 3 + (c - 192)) * NPTS + j] - scen[s][c - 192];
    else v = 0.f;
    X[s][bswz(c, col)] = (_Float16)v;
  }
  __syncthreads();
  int pp = p0 + wave; int b = pp >> 12, n = pp & 4095;
  v8f acc[4];
  wmma_gemm<7, 4>(pwq, X[wave], lane, acc);
  rowmax16<4>(acc);
  if ((lane & 15) == 0) {
    int g = lane >> 4;
#pragma unroll
    for (int t = 0; t < 4; ++t)
#pragma unroll
      for (int v = 0; v < 8; ++v) {
        int c = t * 16 + v + 8 * g;
        float q = tanhf(acc[t][v] + bq[c]);
        size_t o = ((size_t)b * 64 + c) * NPTS + n;
        float zv = zg[o], st = state[o];
        nst[o] = (1.f - zv) * st + zv * q;
      }
  }
}

// ---------------------------------------------------------------------------
// align (SetInterp): gather [feat0, feat1 bcast, dxyz] (259 -> pad 288),
// 259->128 conv + leaky -> 128->64 conv -> softmax over K -> weighted sum of
// gathered next_state. 2 points per 64-thread block (1 wave per point).
// ---------------------------------------------------------------------------
__global__ void align_kernel(const float* __restrict__ feat0,
                             const float* __restrict__ feat1,
                             const float* __restrict__ xyz0,
                             const float* __restrict__ xyz1,
                             const int* __restrict__ idx1,
                             const float* __restrict__ nst,
                             const _Float16* __restrict__ pwa1,
                             const float* __restrict__ ba1,
                             const _Float16* __restrict__ pwa2,
                             const float* __restrict__ ba2,
                             float* __restrict__ out) {
  __shared__ alignas(32) _Float16 X1[2][4608];
  __shared__ alignas(32) _Float16 X2[2][2048];
  __shared__ float GV[2][1024];
  __shared__ float WM[2][1024];
  __shared__ int   sidx[2][16];
  __shared__ float scen[2][3];
  int lane = threadIdx.x & 31, wave = threadIdx.x >> 5;
  int p0 = blockIdx.x * 2;
  if (threadIdx.x < 32) {
    int s = threadIdx.x >> 4, k = threadIdx.x & 15;
    sidx[s][k] = idx1[(size_t)(p0 + s) * KNB + k];
  } else if (threadIdx.x < 38) {
    int r = threadIdx.x - 32; int s = r / 3, c = r % 3;
    int pp = p0 + s;
    scen[s][c] = xyz1[((size_t)(pp >> 12) * 3 + c) * NPTS + (pp & 4095)];
  }
  __syncthreads();
  for (int e = threadIdx.x; e < 2 * 288 * 16; e += 64) {
    int col = e & 15;
    int c = (e >> 4) % 288;
    int s = e / (288 * 16);
    int pp = p0 + s; int b = pp >> 12, n = pp & 4095;
    int j = sidx[s][col];
    float v;
    if (c < 128)      v = feat0[((size_t)b * 128 + c) * NPTS + j];
    else if (c < 256) v = feat1[((size_t)b * 128 + (c - 128)) * NPTS + n];
    else if (c < 259) v = xyz0[((size_t)b * 3 + (c - 256)) * NPTS + j] - scen[s][c - 256];
    else              v = 0.f;
    X1[s][bswz(c, col)] = (_Float16)v;
  }
  for (int e = threadIdx.x; e < 2 * 64 * 16; e += 64) {
    int col = e & 15, c = (e >> 4) & 63, s = e >> 10;
    int pp = p0 + s; int b = pp >> 12;
    int j = sidx[s][col];
    GV[s][c * 16 + col] = nst[((size_t)b * 64 + c) * NPTS + j];
  }
  __syncthreads();
  int s = wave;
  int pp = p0 + s; int b = pp >> 12, n = pp & 4095;
  int col = lane & 15, g = lane >> 4;
  {
    v8f acc[8];
    wmma_gemm<9, 8>(pwa1, X1[s], lane, acc);
#pragma unroll
    for (int t = 0; t < 8; ++t)
#pragma unroll
      for (int v = 0; v < 8; ++v) {
        int c1 = t * 16 + v + 8 * g;
        float x = leaky(acc[t][v] + ba1[c1]);
        X2[s][bswz(c1, col)] = (_Float16)x;
      }
  }
  __syncthreads();
  {
    v8f a2[4];
    wmma_gemm<4, 4>(pwa2, X2[s], lane, a2);
#pragma unroll
    for (int t = 0; t < 4; ++t)
#pragma unroll
      for (int v = 0; v < 8; ++v) {
        int c = t * 16 + v + 8 * g;
        WM[s][c * 16 + col] = a2[t][v] + ba2[c];
      }
  }
  __syncthreads();
#pragma unroll
  for (int h = 0; h < 2; ++h) {
    int c = lane + 32 * h;
    const float* wrow = &WM[s][c * 16];
    const float* grow = &GV[s][c * 16];
    float mx = wrow[0];
#pragma unroll
    for (int k = 1; k < 16; ++k) mx = fmaxf(mx, wrow[k]);
    float sum = 0.f, o = 0.f;
#pragma unroll
    for (int k = 0; k < 16; ++k) {
      float ee = __expf(wrow[k] - mx);
      sum += ee;
      o += grow[k] * ee;
    }
    out[((size_t)b * 64 + c) * NPTS + n] = o / sum;
  }
}

// ---------------------------------------------------------------------------
// host launcher
// ---------------------------------------------------------------------------
extern "C" void kernel_launch(void* const* d_in, const int* in_sizes, int n_in,
                              void* d_out, int out_size, void* d_ws,
                              size_t ws_size, hipStream_t stream) {
  (void)in_sizes; (void)n_in; (void)out_size; (void)ws_size;
  const float* xyz0  = (const float*)d_in[0];
  const float* xyz1  = (const float*)d_in[1];
  const float* state = (const float*)d_in[2];
  const float* corr0 = (const float*)d_in[3];
  const float* feat0 = (const float*)d_in[4];
  const float* feat1 = (const float*)d_in[5];
  const float* flow0 = (const float*)d_in[6];
  const float* fw1 = (const float*)d_in[7];  const float* fb1  = (const float*)d_in[8];
  const float* fg1 = (const float*)d_in[9];  const float* fbe1 = (const float*)d_in[10];
  const float* fw2 = (const float*)d_in[11]; const float* fb2  = (const float*)d_in[12];
  const float* fg2 = (const float*)d_in[13]; const float* fbe2 = (const float*)d_in[14];
  const float* wz  = (const float*)d_in[15]; const float* bz   = (const float*)d_in[16];
  const float* wr  = (const float*)d_in[17]; const float* br_  = (const float*)d_in[18];
  const float* wq  = (const float*)d_in[19]; const float* bq   = (const float*)d_in[20];
  const float* wa1 = (const float*)d_in[21]; const float* ba1  = (const float*)d_in[22];
  const float* wa2 = (const float*)d_in[23]; const float* ba2  = (const float*)d_in[24];

  char* ws = (char*)d_ws;
  size_t off = 0;
  auto carve = [&](size_t bytes) -> char* {
    char* p = ws + off;
    off = (off + bytes + 255) & ~(size_t)255;
    return p;
  };
  _Float16* p_fw1 = (_Float16*)carve((size_t)4 * 1 * 512 * 2);
  _Float16* p_fw2 = (_Float16*)carve((size_t)4 * 2 * 512 * 2);
  _Float16* p_wz  = (_Float16*)carve((size_t)4 * 7 * 512 * 2);
  _Float16* p_wr  = (_Float16*)carve((size_t)4 * 7 * 512 * 2);
  _Float16* p_wq  = (_Float16*)carve((size_t)4 * 7 * 512 * 2);
  _Float16* p_wa1 = (_Float16*)carve((size_t)8 * 9 * 512 * 2);
  _Float16* p_wa2 = (_Float16*)carve((size_t)4 * 4 * 512 * 2);
  int*   idx0   = (int*)carve((size_t)BATCH * NPTS * KNB * 4);
  int*   idx1   = (int*)carve((size_t)BATCH * NPTS * KNB * 4);
  float* stats1 = (float*)carve(8 * 2 * 4);
  float* stats2 = (float*)carve(8 * 2 * 4);
  float* flowf  = (float*)carve((size_t)BATCH * 64 * NPTS * 4);
  float* zg     = (float*)carve((size_t)BATCH * 64 * NPTS * 4);
  float* rg     = (float*)carve((size_t)BATCH * 64 * NPTS * 4);
  float* nst    = (float*)carve((size_t)BATCH * 64 * NPTS * 4);
  float* hbuf   = (float*)carve((size_t)BATCH * NPTS * 64 * KNB * 4);

  auto prep = [&](const float* W, _Float16* P, int Cout, int Cin, int chunks) {
    int total = (Cout / 16) * chunks * 512;
    prep_w_kernel<<<(total + 255) / 256, 256, 0, stream>>>(W, P, Cin, chunks, total);
  };
  prep(fw1, p_fw1, 64, 6, 1);
  prep(fw2, p_fw2, 64, 64, 2);
  prep(wz,  p_wz,  64, 195, 7);
  prep(wr,  p_wr,  64, 195, 7);
  prep(wq,  p_wq,  64, 195, 7);
  prep(wa1, p_wa1, 128, 259, 9);
  prep(wa2, p_wa2, 64, 128, 4);

  knn_kernel<<<(BATCH * NPTS) / 256, 256, 0, stream>>>(xyz0, xyz0, idx0);
  knn_kernel<<<(BATCH * NPTS) / 256, 256, 0, stream>>>(xyz1, xyz0, idx1);

  flow_conv1_kernel<<<BATCH * NPTS / 8, 256, 0, stream>>>(flow0, xyz0, idx0,
                                                          p_fw1, fb1, hbuf);
  gn_stats_kernel<<<8, 256, 0, stream>>>(hbuf, stats1);
  flow_conv2_kernel<<<BATCH * NPTS / 8, 256, 0, stream>>>(hbuf, stats1, fg1,
                                                          fbe1, p_fw2, fb2);
  gn_stats_kernel<<<8, 256, 0, stream>>>(hbuf, stats2);
  flow_final_kernel<<<BATCH * 64 * NPTS / 256, 256, 0, stream>>>(hbuf, stats2,
                                                                 fg2, fbe2, flowf);

  gates_zr_kernel<<<BATCH * NPTS / 8, 256, 0, stream>>>(
      corr0, flowf, state, xyz0, idx0, p_wz, bz, p_wr, br_, zg, rg);
  gate_q_kernel<<<BATCH * NPTS / 8, 256, 0, stream>>>(
      corr0, flowf, state, xyz0, idx0, p_wq, bq, zg, rg, nst);

  align_kernel<<<BATCH * NPTS / 2, 64, 0, stream>>>(
      feat0, feat1, xyz0, xyz1, idx1, nst, p_wa1, ba1, p_wa2, ba2, (float*)d_out);
}
